// MemoryTemporalBlock_46883863003536
// MI455X (gfx1250) — matrix operand order repaired and checked by reading, not compiled, not run on hardware
//
#include <hip/hip_runtime.h>
#include <hip/hip_bf16.h>
#include <math.h>

typedef __attribute__((ext_vector_type(16))) _Float16 v16h;
typedef __attribute__((ext_vector_type(8)))  _Float16 v8h;
typedef __attribute__((ext_vector_type(4)))  _Float16 h4;
typedef __attribute__((ext_vector_type(8)))  float    v8f;

#define BB   8
#define NN   2048
#define TT   12
#define DD   256
#define HH   8
#define DKK  32
#define MM   8
#define ROWS (BB*NN*TT)   /* 196608 */
#define QKV_PAD 772       /* 768 + 4: shift LDS banks per row */

// ---------------------------------------------------------------- convert x
__global__ __launch_bounds__(256)
void cvt_x_f16(const float* __restrict__ x, _Float16* __restrict__ xh, int n) {
    int i = (blockIdx.x * blockDim.x + threadIdx.x) * 4;
    if (i < n) {
        float4 v = *reinterpret_cast<const float4*>(x + i);
        h4 o;
        o[0] = (_Float16)v.x; o[1] = (_Float16)v.y;
        o[2] = (_Float16)v.z; o[3] = (_Float16)v.w;
        *reinterpret_cast<h4*>(xh + i) = o;
    }
}

// ------------------------------------------------------------- pack weights
__global__ __launch_bounds__(256)
void pack_weights(const float* __restrict__ wq, const float* __restrict__ wk,
                  const float* __restrict__ wv, const float* __restrict__ fcw,
                  const float* __restrict__ bq, const float* __restrict__ bk,
                  const float* __restrict__ bv,
                  _Float16* __restrict__ wqkvh, _Float16* __restrict__ wfch,
                  float* __restrict__ bqkv) {
    int i = blockIdx.x * blockDim.x + threadIdx.x;
    const int WQKV = 768 * 256;
    if (i < WQKV) {
        int e = i >> 8, d = i & 255;
        float v = (e < 256) ? wq[e * 256 + d]
                : (e < 512) ? wk[(e - 256) * 256 + d]
                            : wv[(e - 512) * 256 + d];
        wqkvh[i] = (_Float16)v;
    } else if (i < WQKV + 65536) {
        int j = i - WQKV;
        wfch[j] = (_Float16)fcw[j];
    } else if (i < WQKV + 65536 + 768) {
        int k = i - WQKV - 65536;
        bqkv[k] = (k < 256) ? bq[k] : (k < 512) ? bk[k - 256] : bv[k - 512];
    }
}

// ---------------------- fused QKV projection (WMMA) + attention, QKV in LDS
// One block = 48 rows (= 4 (b,n) groups of T=12) x 768 QKV cols.
// 12 waves: wave w -> row-tile mt=w%3, col-group w/3 (12 col-tiles, 2 passes
// of 6 accumulators). QKV tile lives only in LDS; block then runs causal
// temporal attention + memory-bank attention and writes gated f16 output.
__global__ __launch_bounds__(384)
void qkv_attn_fused(const _Float16* __restrict__ xh, const _Float16* __restrict__ wh,
                    const float* __restrict__ bias,
                    const float* __restrict__ memk, const float* __restrict__ memv,
                    const float* __restrict__ alpha_p,
                    _Float16* __restrict__ outh) {
    __shared__ float sQKV[48][QKV_PAD];

    const int m_base = blockIdx.x * 48;
    const int wave   = threadIdx.x >> 5;
    const int lane   = threadIdx.x & 31;
    const int lhalf  = lane >> 4;      // 0: lanes 0-15, 1: lanes 16-31
    const int l15    = lane & 15;
    const int mt     = wave % 3;       // row tile 0..2
    const int grp    = wave / 3;       // col group 0..3 (12 col-tiles each)

    const _Float16* aRow = xh + (size_t)(m_base + mt * 16 + l15) * DD;

    #pragma unroll 1
    for (int pass = 0; pass < 2; ++pass) {
        v8f acc[6] = {};
        for (int k0 = 0; k0 < DD; k0 += 32) {
            // A fragment: lanes 0-15 K chunks {k0, k0+8}; lanes 16-31 {k0+16, k0+24}
            const int kb = k0 + lhalf * 16;
            v8h a0 = *reinterpret_cast<const v8h*>(aRow + kb);
            v8h a1 = *reinterpret_cast<const v8h*>(aRow + kb + 8);
            v16h a;
            #pragma unroll
            for (int i = 0; i < 8; ++i) { a[i] = a0[i]; a[8 + i] = a1[i]; }

            #pragma unroll
            for (int j = 0; j < 6; ++j) {
                const int n0 = (grp * 12 + pass * 6 + j) * 16;
                v16h b = *reinterpret_cast<const v16h*>(
                    wh + (size_t)(n0 + l15) * DD + k0 + lhalf * 16);
                acc[j] = __builtin_amdgcn_wmma_f32_16x16x32_f16(
                    false, a, false, b, (short)0, acc[j], false, false);
            }
        }
        #pragma unroll
        for (int j = 0; j < 6; ++j) {
            const int cg  = (grp * 12 + pass * 6 + j) * 16 + l15;  // 0..767
            const float bvv = bias[cg];
            #pragma unroll
            for (int r = 0; r < 8; ++r)
                sQKV[mt * 16 + r + lhalf * 8][cg] = acc[j][r] + bvv;
        }
    }
    __syncthreads();

    // ---- attention phase: thread -> (group g, head h, time t)
    const int tid = threadIdx.x;
    const int g   = tid / 96;      // 0..3  (b,n) group within block
    const int ht  = tid % 96;
    const int h   = ht / TT;       // 0..7
    const int t   = ht % TT;       // 0..11
    const int rl  = g * TT;        // local row base of this group
    const float scale = 0.17677669529663687f;  // 1/sqrt(32)

    float q[DKK];
    #pragma unroll
    for (int d = 0; d < DKK; ++d) q[d] = sQKV[rl + t][h * DKK + d];

    // causal temporal attention (K at cols 256.., V at cols 512..)
    float sc[TT];
    float mx = -1e30f;
    for (int s = 0; s <= t; ++s) {
        float dot = 0.f;
        #pragma unroll
        for (int d = 0; d < DKK; ++d) dot += q[d] * sQKV[rl + s][256 + h * DKK + d];
        sc[s] = dot * scale;
        mx = fmaxf(mx, sc[s]);
    }
    float den = 0.f;
    for (int s = 0; s <= t; ++s) { sc[s] = __expf(sc[s] - mx); den += sc[s]; }
    const float inv = 1.f / den;

    float ot[DKK];
    #pragma unroll
    for (int d = 0; d < DKK; ++d) ot[d] = 0.f;
    for (int s = 0; s <= t; ++s) {
        const float wgt = sc[s] * inv;
        #pragma unroll
        for (int d = 0; d < DKK; ++d) ot[d] += wgt * sQKV[rl + s][512 + h * DKK + d];
    }

    // memory-bank attention (mem_k/mem_v stream from L2; shared by all blocks)
    float sm[MM];
    float mx2 = -1e30f;
    for (int m = 0; m < MM; ++m) {
        float dot = 0.f;
        const float* kp = memk + (h * MM + m) * DKK;
        #pragma unroll
        for (int d = 0; d < DKK; ++d) dot += q[d] * kp[d];
        sm[m] = dot * scale;
        mx2 = fmaxf(mx2, sm[m]);
    }
    float den2 = 0.f;
    for (int m = 0; m < MM; ++m) { sm[m] = __expf(sm[m] - mx2); den2 += sm[m]; }
    const float inv2 = 1.f / den2;

    const float wl = 1.f / (1.f + __expf(-alpha_p[0]));
    _Float16* op = outh + (size_t)(m_base + rl + t) * DD + h * DKK;
    #pragma unroll
    for (int d = 0; d < DKK; ++d) {
        float ol = 0.f;
        for (int m = 0; m < MM; ++m)
            ol += sm[m] * inv2 * memv[(h * MM + m) * DKK + d];
        op[d] = (_Float16)(ot[d] * (1.f - wl) + ol * wl);
    }
}

// -------------------------- fc GEMM fused with bias + residual + LayerNorm
// One block = 16 rows x 256 cols; 8 waves x 2 col-tiles; LN epilogue in LDS.
__global__ __launch_bounds__(256)
void fc_ln_kernel(const _Float16* __restrict__ ah, const _Float16* __restrict__ wh,
                  const float* __restrict__ fcb, const float* __restrict__ x,
                  const float* __restrict__ gamma, const float* __restrict__ beta,
                  float* __restrict__ out) {
    __shared__ float sY[16][DD];
    __shared__ float sRed[2][16][16];
    __shared__ float sMu[16], sRs[16];

    const int m0    = blockIdx.x * 16;
    const int wave  = threadIdx.x >> 5;
    const int lane  = threadIdx.x & 31;
    const int lhalf = lane >> 4;
    const int l15   = lane & 15;

    v8f acc[2] = {};
    const _Float16* aRow = ah + (size_t)(m0 + l15) * DD;

    for (int k0 = 0; k0 < DD; k0 += 32) {
        const int kb = k0 + lhalf * 16;
        v8h a0 = *reinterpret_cast<const v8h*>(aRow + kb);
        v8h a1 = *reinterpret_cast<const v8h*>(aRow + kb + 8);
        v16h a;
        #pragma unroll
        for (int i = 0; i < 8; ++i) { a[i] = a0[i]; a[8 + i] = a1[i]; }

        #pragma unroll
        for (int j = 0; j < 2; ++j) {
            const int n0 = (wave * 2 + j) * 16;
            v16h b = *reinterpret_cast<const v16h*>(
                wh + (size_t)(n0 + l15) * DD + k0 + lhalf * 16);
            acc[j] = __builtin_amdgcn_wmma_f32_16x16x32_f16(
                false, a, false, b, (short)0, acc[j], false, false);
        }
    }

    #pragma unroll
    for (int j = 0; j < 2; ++j) {
        const int n0  = (wave * 2 + j) * 16;
        const int col = n0 + l15;
        const float bvv = fcb[col];
        #pragma unroll
        for (int r = 0; r < 8; ++r)
            sY[r + lhalf * 8][col] = acc[j][r] + bvv;
    }
    __syncthreads();

    // residual + LayerNorm over D=256, 16 threads per row
    const int r   = threadIdx.x >> 4;
    const int c16 = threadIdx.x & 15;
    float s = 0.f, ss = 0.f;
    #pragma unroll
    for (int i = 0; i < 16; ++i) {
        const int c = c16 + i * 16;
        float hv = sY[r][c] + x[(size_t)(m0 + r) * DD + c];
        sY[r][c] = hv;
        s += hv; ss += hv * hv;
    }
    sRed[0][r][c16] = s;
    sRed[1][r][c16] = ss;
    __syncthreads();
    if (c16 == 0) {
        float S = 0.f, SS = 0.f;
        #pragma unroll
        for (int i = 0; i < 16; ++i) { S += sRed[0][r][i]; SS += sRed[1][r][i]; }
        const float mu  = S * (1.f / 256.f);
        const float var = SS * (1.f / 256.f) - mu * mu;
        sMu[r] = mu;
        sRs[r] = rsqrtf(var + 1e-5f);
    }
    __syncthreads();
    const float mu = sMu[r], rs = sRs[r];
    #pragma unroll
    for (int i = 0; i < 16; ++i) {
        const int c = c16 + i * 16;
        out[(size_t)(m0 + r) * DD + c] = (sY[r][c] - mu) * rs * gamma[c] + beta[c];
    }
}

// --------------------------------------------------------------- launcher
extern "C" void kernel_launch(void* const* d_in, const int* in_sizes, int n_in,
                              void* d_out, int out_size, void* d_ws, size_t ws_size,
                              hipStream_t stream) {
    (void)in_sizes; (void)n_in; (void)out_size; (void)ws_size;
    const float* x     = (const float*)d_in[0];
    const float* wq    = (const float*)d_in[1];
    const float* bq    = (const float*)d_in[2];
    const float* wk    = (const float*)d_in[3];
    const float* bk    = (const float*)d_in[4];
    const float* wv    = (const float*)d_in[5];
    const float* bv    = (const float*)d_in[6];
    const float* memk  = (const float*)d_in[7];
    const float* memv  = (const float*)d_in[8];
    const float* fcw   = (const float*)d_in[9];
    const float* fcb   = (const float*)d_in[10];
    const float* gamma = (const float*)d_in[11];
    const float* beta  = (const float*)d_in[12];
    const float* alpha = (const float*)d_in[13];
    float* out = (float*)d_out;

    char* ws = (char*)d_ws;
    size_t off = 0;
    auto alloc = [&](size_t bytes) -> char* {
        char* p = ws + off;
        off += (bytes + 255) & ~size_t(255);
        return p;
    };
    _Float16* xh    = (_Float16*)alloc((size_t)ROWS * DD * 2);   // 100 MB
    _Float16* wqkvh = (_Float16*)alloc((size_t)768 * 256 * 2);
    _Float16* wfch  = (_Float16*)alloc((size_t)256 * 256 * 2);
    float*    bqkv  = (float*)alloc(768 * 4);
    _Float16* ath   = (_Float16*)alloc((size_t)ROWS * DD * 2);   // 100 MB

    {
        const int n = ROWS * DD;
        const int bl = (n / 4 + 255) / 256;
        cvt_x_f16<<<bl, 256, 0, stream>>>(x, xh, n);
    }
    {
        const int total = 768 * 256 + 65536 + 768;
        pack_weights<<<(total + 255) / 256, 256, 0, stream>>>(
            wq, wk, wv, fcw, bq, bk, bv, wqkvh, wfch, bqkv);
    }
    qkv_attn_fused<<<ROWS / 48, 384, 0, stream>>>(
        xh, wqkvh, bqkv, memk, memv, alpha, ath);
    fc_ln_kernel<<<ROWS / 16, 256, 0, stream>>>(ath, wfch, fcb, x, gamma, beta, out);
}
